// GATClassifier_45861660787101
// MI455X (gfx1250) — matrix operand order repaired and checked by reading, not compiled, not run on hardware
//
#include <hip/hip_runtime.h>

#define N_NODES 50000
#define N_EDGES 800000
#define N_GRAPHS 256
#define IN_DIM 128
#define HID 64
#define HEADS 4
#define NEG_SLOPE 0.2f

typedef __bf16 bf16;
typedef __attribute__((ext_vector_type(16))) __bf16 v16bf;
typedef __attribute__((ext_vector_type(8)))  __bf16 v8bf;
typedef __attribute__((ext_vector_type(8)))  float  v8f;

// ---------------- helpers ----------------

__device__ __forceinline__ unsigned encf(float f) {
    unsigned u = __float_as_uint(f);
    return (u & 0x80000000u) ? ~u : (u | 0x80000000u);
}
__device__ __forceinline__ float decf(unsigned e) {
    unsigned u = (e & 0x80000000u) ? (e & 0x7fffffffu) : ~e;
    return __uint_as_float(u);
}
__device__ __forceinline__ float lrelu(float v) {
    return v > 0.f ? v : NEG_SLOPE * v;
}

// ---------------- conversion / packing ----------------

__global__ void cvt_f32_bf16(const float* __restrict__ in, bf16* __restrict__ out, int n) {
    int tid = blockIdx.x * blockDim.x + threadIdx.x;
    if (tid < n) out[tid] = (bf16)in[tid];
}

// Pack row-major W[K,N] (f32) into per-wave WMMA B fragments (bf16).
// Fragment (kt, nt): lane holds B[kt*32 + (lane>=16?8:0) + (j&7) + (j>=8?16:0), nt*16 + (lane&15)], j=0..15
__global__ void pack_w_bfrag(const float* __restrict__ W, bf16* __restrict__ out, int K, int N) {
    int tid = blockIdx.x * blockDim.x + threadIdx.x;
    int NT = N >> 4, KT = K >> 5;
    if (tid >= NT * KT * 32) return;
    int lane = tid & 31, frag = tid >> 5;
    int kt = frag / NT, nt = frag - kt * NT;
    int col = (nt << 4) + (lane & 15);
    int kb  = (kt << 5) + ((lane >> 4) << 3);
    bf16* o = out + ((size_t)frag << 9) + (lane << 4);
    for (int j = 0; j < 16; ++j)
        o[j] = (bf16)W[(size_t)(kb + (j & 7) + ((j >> 3) << 4)) * N + col];
}

// ---------------- WMMA GEMM: C[M,N] = A[M,K] @ B[K,N] ----------------
// A row-major bf16, B pre-packed fragments, C row-major f32.
// grid = (M/16, NT/4), block = 128 (4 waves, one 16x16 tile per wave).

__global__ void __launch_bounds__(128)
gemm_bf16_wmma(const bf16* __restrict__ A, const bf16* __restrict__ Bp,
               float* __restrict__ C, int K, int N) {
    const int lane  = threadIdx.x & 31;
    const int wave  = threadIdx.x >> 5;
    const int NT    = N >> 4;
    const int ntile = blockIdx.y * 4 + wave;
    const int row0  = blockIdx.x << 4;

    v8f acc = {0.f, 0.f, 0.f, 0.f, 0.f, 0.f, 0.f, 0.f};

    // A fragment addressing: lane<16 -> row=lane, K chunks [0..7] & [16..23];
    // lane>=16 -> same rows, K chunks [8..15] & [24..31].
    const bf16* arow = A + (size_t)(row0 + (lane & 15)) * K + ((lane >> 4) << 3);
    const int kiter = K >> 5;
    for (int kt = 0; kt < kiter; ++kt) {
        const bf16* pa = arow + (kt << 5);
        v8bf lo = *(const v8bf*)(pa);
        v8bf hi = *(const v8bf*)(pa + 16);
        v16bf a;
#pragma unroll
        for (int i = 0; i < 8; ++i) { a[i] = lo[i]; a[i + 8] = hi[i]; }
        v16bf b = *(const v16bf*)(Bp + (((size_t)kt * NT + ntile) << 9) + (lane << 4));
        acc = __builtin_amdgcn_wmma_f32_16x16x32_bf16(
            false, a, false, b, (short)0, acc, false, false);
    }

    // C layout: VGPR j, lane -> M = (lane>=16?8:0)+j, N = lane&15
    float* crow = C + (size_t)(row0 + ((lane >> 4) << 3)) * N + (ntile << 4) + (lane & 15);
#pragma unroll
    for (int j = 0; j < 8; ++j) crow[(size_t)j * N] = acc[j];
}

// ---------------- attention scalar projections ----------------
// el/er[node*H+h] = sum_d feat[node,h,d] * attn[h,d]

__global__ void el_er_k(const float* __restrict__ feat, const float* __restrict__ al,
                        const float* __restrict__ ar, float* __restrict__ el,
                        float* __restrict__ er, int n, int H) {
    int tid = blockIdx.x * blockDim.x + threadIdx.x;
    if (tid >= n * H) return;
    int head = tid % H;
    const float* f = feat + (size_t)tid * HID;
    const float* a = al + head * HID;
    const float* b = ar + head * HID;
    float sl = 0.f, sr = 0.f;
    for (int d = 0; d < HID; ++d) { float v = f[d]; sl += v * a[d]; sr += v * b[d]; }
    el[tid] = sl; er[tid] = sr;
}

// ---------------- edge softmax passes ----------------

template <int H>
__global__ void edge_max_k(const int* __restrict__ src, const int* __restrict__ dst,
                           const float* __restrict__ el, const float* __restrict__ er,
                           unsigned* __restrict__ emax, int E) {
    int tid = blockIdx.x * blockDim.x + threadIdx.x;
    if (tid >= E * H) return;
    int e = tid / H, h = tid % H;
    int s = src[e], d = dst[e];
    float v = lrelu(el[s * H + h] + er[d * H + h]);
    atomicMax(emax + d * H + h, encf(v));
}

template <int H>
__global__ void edge_sum_k(const int* __restrict__ src, const int* __restrict__ dst,
                           const float* __restrict__ el, const float* __restrict__ er,
                           const unsigned* __restrict__ emax, float* __restrict__ denom, int E) {
    int tid = blockIdx.x * blockDim.x + threadIdx.x;
    if (tid >= E * H) return;
    int e = tid / H, h = tid % H;
    int s = src[e], d = dst[e];
    float v = lrelu(el[s * H + h] + er[d * H + h]);
    float ee = __expf(v - decf(emax[d * H + h]));
    atomicAdd(denom + d * H + h, ee);
}

// rst[dst, h, d] += alpha(edge, h) * feat[src, h, d]
template <int H>
__global__ void __launch_bounds__(256)
aggregate_k(const int* __restrict__ src, const int* __restrict__ dst,
            const float* __restrict__ el, const float* __restrict__ er,
            const unsigned* __restrict__ emax, const float* __restrict__ denom,
            const float* __restrict__ feat, float* __restrict__ rst, int E) {
    const int HD  = H * HID;
    const int PER = 256 / HD;       // edges handled concurrently per block
    const int EPB = 64;             // edges per block
    int sub = threadIdx.x / HD;
    int t   = threadIdx.x % HD;
    int h   = t >> 6;
    int base = blockIdx.x * EPB;
    for (int i = sub; i < EPB; i += PER) {
        int e = base + i;
        if (e >= E) continue;
        int s = src[e], d = dst[e];
        float v  = lrelu(el[s * H + h] + er[d * H + h]);
        float ee = __expf(v - decf(emax[d * H + h]));
        float alpha = ee / fmaxf(denom[d * H + h], 1e-9f);
        atomicAdd(rst + (size_t)d * HD + t, alpha * feat[(size_t)s * HD + t]);
    }
}

// ---------------- epilogues ----------------

__global__ void relu_bias_cvt(const float* __restrict__ rst, const float* __restrict__ bias,
                              bf16* __restrict__ xb, int n, int C) {
    int tid = blockIdx.x * blockDim.x + threadIdx.x;
    if (tid >= n) return;
    float v = rst[tid] + bias[tid % C];
    xb[tid] = (bf16)fmaxf(v, 0.f);
}

__global__ void pool_k(const float* __restrict__ rst2, const float* __restrict__ b2,
                       const int* __restrict__ gid, float* __restrict__ sums,
                       float* __restrict__ cnt, int n) {
    int tid = blockIdx.x * blockDim.x + threadIdx.x;
    if (tid >= n) return;
    int node = tid >> 6, d = tid & 63;
    float v = fmaxf(rst2[tid] + b2[d], 0.f);
    int g = gid[node];
    atomicAdd(sums + g * HID + d, v);
    if (d == 0) atomicAdd(cnt + g, 1.0f);
}

__global__ void classify_k(const float* __restrict__ sums, const float* __restrict__ cnt,
                           const float* __restrict__ Wc, const float* __restrict__ bc,
                           float* __restrict__ out) {
    int tid = threadIdx.x;           // 512 = 256 graphs x 2 classes
    int g = tid >> 1, c = tid & 1;
    float inv = 1.0f / fmaxf(cnt[g], 1.0f);
    float acc = bc[c];
    for (int d = 0; d < HID; ++d)
        acc += sums[g * HID + d] * inv * Wc[d * 2 + c];
    out[tid] = acc;
}

// ---------------- host launcher ----------------

extern "C" void kernel_launch(void* const* d_in, const int* in_sizes, int n_in,
                              void* d_out, int out_size, void* d_ws, size_t ws_size,
                              hipStream_t stream) {
    const float* h    = (const float*)d_in[0];
    const int*   src  = (const int*)d_in[1];
    const int*   dst  = (const int*)d_in[2];
    const int*   gid  = (const int*)d_in[3];
    const float* W1   = (const float*)d_in[4];
    const float* al1  = (const float*)d_in[5];
    const float* ar1  = (const float*)d_in[6];
    const float* b1   = (const float*)d_in[7];
    const float* W2   = (const float*)d_in[8];
    const float* al2  = (const float*)d_in[9];
    const float* ar2  = (const float*)d_in[10];
    const float* b2   = (const float*)d_in[11];
    const float* Wc   = (const float*)d_in[12];
    const float* bc   = (const float*)d_in[13];
    float* out = (float*)d_out;

    char* ws = (char*)d_ws;
    size_t off = 0;
    auto alloc = [&](size_t bytes) -> char* {
        char* p = ws + off;
        off = (off + bytes + 255) & ~(size_t)255;
        return p;
    };

    bf16*     h_bf   = (bf16*)alloc((size_t)N_NODES * IN_DIM * 2);           // 12.8 MB
    bf16*     W1p    = (bf16*)alloc((size_t)IN_DIM * HEADS * HID * 2);
    bf16*     W2p    = (bf16*)alloc((size_t)HEADS * HID * HID * 2);
    float*    feat1  = (float*)alloc((size_t)N_NODES * HEADS * HID * 4);     // 51.2 MB
    float*    el1    = (float*)alloc((size_t)N_NODES * HEADS * 4);
    float*    er1    = (float*)alloc((size_t)N_NODES * HEADS * 4);
    unsigned* emax1  = (unsigned*)alloc((size_t)N_NODES * HEADS * 4);
    float*    den1   = (float*)alloc((size_t)N_NODES * HEADS * 4);
    float*    rst1   = (float*)alloc((size_t)N_NODES * HEADS * HID * 4);     // 51.2 MB
    bf16*     x_bf   = (bf16*)alloc((size_t)N_NODES * HEADS * HID * 2);      // 25.6 MB
    float*    el2    = (float*)alloc((size_t)N_NODES * 4);
    float*    er2    = (float*)alloc((size_t)N_NODES * 4);
    unsigned* emax2  = (unsigned*)alloc((size_t)N_NODES * 4);
    float*    den2   = (float*)alloc((size_t)N_NODES * 4);
    float*    sums   = (float*)alloc((size_t)N_GRAPHS * HID * 4);
    float*    cnt    = (float*)alloc((size_t)N_GRAPHS * 4);
    // layer-2 buffers alias dead layer-1 buffers (written fully before reads)
    float*    feat2  = feat1;    // [N, 64] f32 (dead after aggregate<4>)
    float*    rst2   = rst1;     // [N, 64] f32 (dead after relu_bias_cvt)

    const int HD1 = HEADS * HID;   // 256

    // ---- weight pack + input conversion ----
    pack_w_bfrag<<<(64 * 32 + 255) / 256, 256, 0, stream>>>(W1, W1p, IN_DIM, HD1);
    pack_w_bfrag<<<(32 * 32 + 255) / 256, 256, 0, stream>>>(W2, W2p, HD1, HID);
    cvt_f32_bf16<<<(N_NODES * IN_DIM + 255) / 256, 256, 0, stream>>>(h, h_bf, N_NODES * IN_DIM);

    // ---- layer 1 ----
    hipMemsetAsync(emax1, 0, (size_t)N_NODES * HEADS * 4, stream);
    hipMemsetAsync(den1,  0, (size_t)N_NODES * HEADS * 4, stream);
    hipMemsetAsync(rst1,  0, (size_t)N_NODES * HD1 * 4, stream);

    gemm_bf16_wmma<<<dim3(N_NODES / 16, (HD1 / 16) / 4), 128, 0, stream>>>(
        h_bf, W1p, feat1, IN_DIM, HD1);
    el_er_k<<<(N_NODES * HEADS + 255) / 256, 256, 0, stream>>>(
        feat1, al1, ar1, el1, er1, N_NODES, HEADS);
    edge_max_k<HEADS><<<(N_EDGES * HEADS + 255) / 256, 256, 0, stream>>>(
        src, dst, el1, er1, emax1, N_EDGES);
    edge_sum_k<HEADS><<<(N_EDGES * HEADS + 255) / 256, 256, 0, stream>>>(
        src, dst, el1, er1, emax1, den1, N_EDGES);
    aggregate_k<HEADS><<<(N_EDGES + 63) / 64, 256, 0, stream>>>(
        src, dst, el1, er1, emax1, den1, feat1, rst1, N_EDGES);
    relu_bias_cvt<<<(N_NODES * HD1 + 255) / 256, 256, 0, stream>>>(
        rst1, b1, x_bf, N_NODES * HD1, HD1);

    // ---- layer 2 (rst2/feat2 alias now-dead layer-1 buffers) ----
    hipMemsetAsync(emax2, 0, (size_t)N_NODES * 4, stream);
    hipMemsetAsync(den2,  0, (size_t)N_NODES * 4, stream);
    hipMemsetAsync(rst2,  0, (size_t)N_NODES * HID * 4, stream);
    hipMemsetAsync(sums,  0, (size_t)N_GRAPHS * HID * 4, stream);
    hipMemsetAsync(cnt,   0, (size_t)N_GRAPHS * 4, stream);

    gemm_bf16_wmma<<<dim3(N_NODES / 16, 1), 128, 0, stream>>>(
        x_bf, W2p, feat2, HD1, HID);
    el_er_k<<<(N_NODES + 255) / 256, 256, 0, stream>>>(
        feat2, al2, ar2, el2, er2, N_NODES, 1);
    edge_max_k<1><<<(N_EDGES + 255) / 256, 256, 0, stream>>>(
        src, dst, el2, er2, emax2, N_EDGES);
    edge_sum_k<1><<<(N_EDGES + 255) / 256, 256, 0, stream>>>(
        src, dst, el2, er2, emax2, den2, N_EDGES);
    aggregate_k<1><<<(N_EDGES + 63) / 64, 256, 0, stream>>>(
        src, dst, el2, er2, emax2, den2, feat2, rst2, N_EDGES);

    // ---- pool + classify ----
    pool_k<<<(N_NODES * HID + 255) / 256, 256, 0, stream>>>(
        rst2, b2, gid, sums, cnt, N_NODES * HID);
    classify_k<<<1, N_GRAPHS * 2, 0, stream>>>(sums, cnt, Wc, bc, out);
}
